// PointNetSetAbstraction_87024627352064
// MI455X (gfx1250) — matrix operand order repaired and checked by reading, not compiled, not run on hardware
//
#include <hip/hip_runtime.h>

// ---------------- problem constants ----------------
#define BATCH  16
#define NPTS   4096
#define DFEAT  64
#define SPTS   1024      // NPOINT
#define KSAMP  32        // NSAMPLE
#define R2     0.04f     // RADIUS^2
#define MROWS  (BATCH * SPTS * KSAMP)   // 524288 rows of the "im2col" matrix
#define INV_M  (1.0f / (float)MROWS)

// WMMA vector types (CDNA5 wave32, V_WMMA_F32_16X16X32_F16)
typedef __attribute__((ext_vector_type(16))) _Float16 v16h;
typedef __attribute__((ext_vector_type(8)))  float    v8f;

union Frag  { uint4 u[2]; v16h h; };
union Pack8 { uint4 u; _Float16 h[8]; };

// =====================================================================
// K0: convert weights fp32 [Cin][Cout] -> fp16 transposed [Cout][Kpad]
//     w0: 67x64 -> [64][96] (zero pad k>=67), w1: 64x64 -> [64][64],
//     w2: 64x128 -> [128][64]
// =====================================================================
__global__ void prep_weights(const float* __restrict__ w0,
                             const float* __restrict__ w1,
                             const float* __restrict__ w2,
                             _Float16* __restrict__ wt) {
  int i = blockIdx.x * 256 + threadIdx.x;
  if (i < 64 * 96) {
    int n = i / 96, k = i % 96;
    wt[i] = (_Float16)(k < 67 ? w0[k * 64 + n] : 0.0f);
  } else if (i < 64 * 96 + 64 * 64) {
    int j = i - 64 * 96; int n = j / 64, k = j % 64;
    wt[8192 + j] = (_Float16)w1[k * 64 + n];
  } else if (i < 64 * 96 + 64 * 64 + 128 * 64) {
    int j = i - (64 * 96 + 64 * 64); int n = j / 64, k = j % 64;
    wt[16384 + j] = (_Float16)w2[k * 128 + n];
  }
}

// K0b: zero the BN statistic accumulators (3 layers x 512 floats)
__global__ void init_stats(float* __restrict__ stat) {
  int i = blockIdx.x * 256 + threadIdx.x;
  if (i < 3 * 512) stat[i] = 0.0f;
}

// =====================================================================
// K1: farthest point sampling. One workgroup per batch, whole cloud in
//     LDS, per-thread distance array in registers, 2-stage argmax
//     (wave shfl reduce + 8-entry LDS reduce). Writes new_xyz [B][S][3]
//     and the transposed new_xyz output [B][3][S].
// =====================================================================
__global__ void __launch_bounds__(256) fps_kernel(const float* __restrict__ xyz,
                                                  float* __restrict__ nxyz,
                                                  float* __restrict__ out_xyz) {
  __shared__ float sx[NPTS], sy[NPTS], sz[NPTS];
  __shared__ float rbest[8];
  __shared__ int   ribest[8];
  __shared__ int   sfar;

  const int b   = blockIdx.x;
  const int tid = threadIdx.x;
  const int lane = tid & 31, wave = tid >> 5;
  const float* xb = xyz + (long)b * 3 * NPTS;

  for (int i = tid; i < NPTS; i += 256) {
    sx[i] = xb[i];
    sy[i] = xb[NPTS + i];
    sz[i] = xb[2 * NPTS + i];
  }
  if (tid == 0) sfar = 0;

  float dreg[16];                     // distances for points i = w*256+tid
  #pragma unroll
  for (int w = 0; w < 16; w++) dreg[w] = 1e10f;
  __syncthreads();

  for (int it = 0; it < SPTS; it++) {
    const int far = sfar;
    const float cx = sx[far], cy = sy[far], cz = sz[far];
    if (tid == 0) {
      nxyz[((long)b * SPTS + it) * 3 + 0] = cx;
      nxyz[((long)b * SPTS + it) * 3 + 1] = cy;
      nxyz[((long)b * SPTS + it) * 3 + 2] = cz;
      out_xyz[((long)b * 3 + 0) * SPTS + it] = cx;
      out_xyz[((long)b * 3 + 1) * SPTS + it] = cy;
      out_xyz[((long)b * 3 + 2) * SPTS + it] = cz;
    }
    float bv = -1.0f; int bi = 0x7fffffff;
    #pragma unroll
    for (int w = 0; w < 16; w++) {
      const int i = w * 256 + tid;
      const float dx = sx[i] - cx, dy = sy[i] - cy, dz = sz[i] - cz;
      const float d = dx * dx + dy * dy + dz * dz;
      const float dm = fminf(dreg[w], d);
      dreg[w] = dm;
      if (dm > bv) { bv = dm; bi = i; }           // i ascending -> first max
    }
    // wave32 argmax reduce (tie -> lowest index, matching jnp.argmax)
    #pragma unroll
    for (int off = 16; off > 0; off >>= 1) {
      const float ov = __shfl_down(bv, off);
      const int   oi = __shfl_down(bi, off);
      if (ov > bv || (ov == bv && oi < bi)) { bv = ov; bi = oi; }
    }
    if (lane == 0) { rbest[wave] = bv; ribest[wave] = bi; }
    __syncthreads();
    if (tid == 0) {
      float mb = rbest[0]; int mi = ribest[0];
      #pragma unroll
      for (int w = 1; w < 8; w++)
        if (rbest[w] > mb || (rbest[w] == mb && ribest[w] < mi)) { mb = rbest[w]; mi = ribest[w]; }
      sfar = mi;
    }
    __syncthreads();
  }
}

// =====================================================================
// K2: ball query. One wave32 per centroid; ballot/popc prefix append
//     reproduces "sorted index, first 32 within radius, pad with first"
// =====================================================================
__global__ void __launch_bounds__(256) query_ball(const float* __restrict__ xyz,
                                                  const float* __restrict__ nxyz,
                                                  int* __restrict__ idx) {
  const int tid = threadIdx.x, lane = tid & 31, wave = tid >> 5;
  const long u = (long)blockIdx.x * 8 + wave;     // b*SPTS + s
  const int b = (int)(u >> 10);
  const float* xb = xyz + (long)b * 3 * NPTS;
  const float cx = nxyz[u * 3 + 0], cy = nxyz[u * 3 + 1], cz = nxyz[u * 3 + 2];
  int* ob = idx + u * KSAMP;
  const unsigned lmask = (1u << lane) - 1u;

  int cnt = 0, firstp = -1;
  for (int base = 0; base < NPTS && cnt < KSAMP; base += 32) {
    const int i = base + lane;
    const float dx = xb[i] - cx, dy = xb[NPTS + i] - cy, dz = xb[2 * NPTS + i] - cz;
    const bool in = (dx * dx + dy * dy + dz * dz) <= R2;
    const unsigned m = (unsigned)__ballot(in);
    if (firstp < 0 && m) firstp = base + __ffs(m) - 1;
    const int pos = cnt + __popc(m & lmask);
    if (in && pos < KSAMP) ob[pos] = i;
    cnt += __popc(m);
  }
  if (cnt < KSAMP) {
    const int pv = (firstp >= 0) ? firstp : (NPTS - 1);
    if (lane >= cnt) ob[lane] = pv;
  }
}

// =====================================================================
// K3: gather + concat -> h0 [MROWS][96] fp16 (cols 0-2 centered xyz,
//     3-66 point features, 67-95 zero pad). One wave per row.
// =====================================================================
__global__ void __launch_bounds__(256) build_h0(const float* __restrict__ xyz,
                                                const float* __restrict__ pts,
                                                const int* __restrict__ idx,
                                                const float* __restrict__ nxyz,
                                                _Float16* __restrict__ h0) {
  const int tid = threadIdx.x, lane = tid & 31, wave = tid >> 5;
  const long r = (long)blockIdx.x * 8 + wave;     // row in [0, MROWS)
  const int b = (int)(r >> 15);
  const int s = (int)((r >> 5) & (SPTS - 1));
  const int j = idx[r];
  _Float16* o = h0 + r * 96;

  float v;
  if (lane < 3)
    v = xyz[((long)b * 3 + lane) * NPTS + j] - nxyz[((long)b * SPTS + s) * 3 + lane];
  else
    v = pts[((long)b * DFEAT + (lane - 3)) * NPTS + j];
  o[lane] = (_Float16)v;

  v = pts[((long)b * DFEAT + (lane + 29)) * NPTS + j];
  o[lane + 32] = (_Float16)v;

  v = (lane < 3) ? pts[((long)b * DFEAT + (lane + 61)) * NPTS + j] : 0.0f;
  o[lane + 64] = (_Float16)v;
}

// =====================================================================
// K4: WMMA GEMM  Y[M][ND] = A[M][KD] * W[KD][ND] + bias, fp16 in,
//     f32 accumulate, fp16 out; fused per-channel sum/sum^2 for BN.
//     Each wave processes MS consecutive 16-row strips so the B
//     fragments (weights) are loaded once per k-step and reused MS
//     times -> MS*NT independent WMMAs per k-step hide load latency.
//     WT is the transposed fp16 weight [ND][KD].
// =====================================================================
template <int KD, int ND, int MS>
__global__ void __launch_bounds__(256) gemm_wmma(const _Float16* __restrict__ A,
                                                 const _Float16* __restrict__ WT,
                                                 const float* __restrict__ bias,
                                                 _Float16* __restrict__ Y,
                                                 float* __restrict__ gsum,
                                                 float* __restrict__ gsq) {
  __shared__ float lsum[ND];
  __shared__ float lsq[ND];
  const int tid = threadIdx.x;
  for (int i = tid; i < ND; i += 256) { lsum[i] = 0.0f; lsq[i] = 0.0f; }
  __syncthreads();

  const int wave = tid >> 5, lane = tid & 31;
  const int row = lane & 15, hig = lane >> 4;     // hig=0: lanes 0-15, 1: 16-31
  const long strip0 = ((long)blockIdx.x * 8 + wave) * MS;

  constexpr int NT = ND / 16;
  v8f acc[MS][NT];
  #pragma unroll
  for (int m = 0; m < MS; m++)
    #pragma unroll
    for (int t = 0; t < NT; t++)
      #pragma unroll
      for (int e = 0; e < 8; e++) acc[m][t][e] = 0.0f;

  #pragma unroll
  for (int kc = 0; kc < KD; kc += 32) {
    // B fragments: column n = t*16+row, 16 contiguous k at kc + hig*16
    Frag bf[NT];
    #pragma unroll
    for (int t = 0; t < NT; t++) {
      const _Float16* bp = WT + ((long)(t * 16 + row)) * KD + kc + hig * 16;
      bf[t].u[0] = *reinterpret_cast<const uint4*>(bp);
      bf[t].u[1] = *reinterpret_cast<const uint4*>(bp + 8);
    }
    #pragma unroll
    for (int m = 0; m < MS; m++) {
      // A fragment: 16 contiguous k at kc + hig*16 (matches B)
      // halves 8-15 = +8
      const _Float16* arow = A + ((strip0 + m) * 16 + row) * KD + hig * 16;
      Frag af;
      af.u[0] = *reinterpret_cast<const uint4*>(arow + kc);
      af.u[1] = *reinterpret_cast<const uint4*>(arow + kc + 8);
      #pragma unroll
      for (int t = 0; t < NT; t++)
        acc[m][t] = __builtin_amdgcn_wmma_f32_16x16x32_f16(
            false, af.h, false, bf[t].h, (short)0, acc[m][t], false, false);
    }
  }

  // epilogue: +bias, store fp16, accumulate BN stats
  #pragma unroll
  for (int m = 0; m < MS; m++) {
    #pragma unroll
    for (int t = 0; t < NT; t++) {
      const int n = t * 16 + row;
      const float bv = bias[n];
      _Float16* yp = Y + ((strip0 + m) * 16 + hig * 8) * ND + n;
      float ps = 0.0f, pq = 0.0f;
      #pragma unroll
      for (int rr = 0; rr < 8; rr++) {          // D layout: m = strip*16 + hig*8 + rr
        const float v = acc[m][t][rr] + bv;
        ps += v; pq += v * v;
        yp[(long)rr * ND] = (_Float16)v;
      }
      atomicAdd(&lsum[n], ps);
      atomicAdd(&lsq[n], pq);
    }
  }
  __syncthreads();
  for (int i = tid; i < ND; i += 256) {
    atomicAdd(&gsum[i], lsum[i]);
    atomicAdd(&gsq[i], lsq[i]);
  }
}

// K5: fold BN stats into per-channel scale/shift
__global__ void finalize_stats(const float* __restrict__ sum,
                               const float* __restrict__ sumsq,
                               const float* __restrict__ g,
                               const float* __restrict__ be,
                               float* __restrict__ scale,
                               float* __restrict__ shift, int nd) {
  const int n = threadIdx.x;
  if (n < nd) {
    const float mu  = sum[n] * INV_M;
    const float var = sumsq[n] * INV_M - mu * mu;
    const float sc  = g[n] * rsqrtf(var + 1e-5f);
    scale[n] = sc;
    shift[n] = be[n] - mu * sc;
  }
}

// K6: in-place BN + ReLU on fp16 activations, ND = 64.
//     Vectorized: each thread handles 8 halves with b128 load/store.
__global__ void bn_relu64(_Float16* __restrict__ y,
                          const float* __restrict__ scale,
                          const float* __restrict__ shift) {
  const long i = (long)blockIdx.x * 256 + threadIdx.x;  // 8-half packet
  const int c0 = (int)((i * 8) & 63);
  const float4 s0 = *reinterpret_cast<const float4*>(scale + c0);
  const float4 s1 = *reinterpret_cast<const float4*>(scale + c0 + 4);
  const float4 h0 = *reinterpret_cast<const float4*>(shift + c0);
  const float4 h1 = *reinterpret_cast<const float4*>(shift + c0 + 4);
  Pack8 p;
  p.u = *reinterpret_cast<const uint4*>(y + i * 8);
  p.h[0] = (_Float16)fmaxf((float)p.h[0] * s0.x + h0.x, 0.0f);
  p.h[1] = (_Float16)fmaxf((float)p.h[1] * s0.y + h0.y, 0.0f);
  p.h[2] = (_Float16)fmaxf((float)p.h[2] * s0.z + h0.z, 0.0f);
  p.h[3] = (_Float16)fmaxf((float)p.h[3] * s0.w + h0.w, 0.0f);
  p.h[4] = (_Float16)fmaxf((float)p.h[4] * s1.x + h1.x, 0.0f);
  p.h[5] = (_Float16)fmaxf((float)p.h[5] * s1.y + h1.y, 0.0f);
  p.h[6] = (_Float16)fmaxf((float)p.h[6] * s1.z + h1.z, 0.0f);
  p.h[7] = (_Float16)fmaxf((float)p.h[7] * s1.w + h1.w, 0.0f);
  *reinterpret_cast<uint4*>(y + i * 8) = p.u;
}

// K7: BN + ReLU + max over K=32, write new_points [B][128][S].
//     One wave per (b,s); lane handles channels c, c+32, c+64, c+96.
__global__ void __launch_bounds__(256) final_max(const _Float16* __restrict__ y2,
                                                 const float* __restrict__ scale,
                                                 const float* __restrict__ shift,
                                                 float* __restrict__ out) {
  const int tid = threadIdx.x, lane = tid & 31, wave = tid >> 5;
  const long u = (long)blockIdx.x * 8 + wave;     // b*SPTS + s
  const int s = (int)(u & (SPTS - 1));
  const int b = (int)(u >> 10);
  const _Float16* base = y2 + u * KSAMP * 128;
  #pragma unroll
  for (int cc = 0; cc < 4; cc++) {
    const int c = lane + cc * 32;
    const float sc = scale[c], sh = shift[c];
    float m = 0.0f;                               // ReLU -> all vals >= 0
    #pragma unroll 4
    for (int k = 0; k < KSAMP; k++) {
      const float v = fmaxf((float)base[(long)k * 128 + c] * sc + sh, 0.0f);
      m = fmaxf(m, v);
    }
    out[((long)b * 128 + c) * SPTS + s] = m;
  }
}

// =====================================================================
// launch
// =====================================================================
extern "C" void kernel_launch(void* const* d_in, const int* in_sizes, int n_in,
                              void* d_out, int out_size, void* d_ws, size_t ws_size,
                              hipStream_t stream) {
  (void)in_sizes; (void)n_in; (void)out_size; (void)ws_size;
  const float* xyz = (const float*)d_in[0];
  const float* pts = (const float*)d_in[1];
  const float* w0  = (const float*)d_in[2];
  const float* b0  = (const float*)d_in[3];
  const float* g0  = (const float*)d_in[4];
  const float* be0 = (const float*)d_in[5];
  const float* w1  = (const float*)d_in[6];
  const float* b1  = (const float*)d_in[7];
  const float* g1  = (const float*)d_in[8];
  const float* be1 = (const float*)d_in[9];
  const float* w2  = (const float*)d_in[10];
  const float* b2  = (const float*)d_in[11];
  const float* g2  = (const float*)d_in[12];
  const float* be2 = (const float*)d_in[13];
  float* out = (float*)d_out;

  // workspace layout (~294 MB total, buffers aliased across stages)
  char* ws = (char*)d_ws;
  int*      idx  = (int*)(ws);                            // 2 MB: [B*S*K]
  float*    nxyz = (float*)(ws + (3l << 20));             // [B*S*3] f32
  float*    stat = (float*)(ws + (4l << 20));             // 3 layers x 512 f32
  _Float16* wt   = (_Float16*)(ws + (5l << 20));          // fp16 transposed weights
  _Float16* y0   = (_Float16*)(ws + (6l << 20));          // [M][64]  (-> h1 in place)
  _Float16* y2   = (_Float16*)(ws + (70l << 20));         // [M][128]
  _Float16* h0   = (_Float16*)(ws + (198l << 20));        // [M][96]
  _Float16* y1   = h0;                                    // reuse after GEMM0 (-> h2)

  prep_weights<<<72, 256, 0, stream>>>(w0, w1, w2, wt);
  init_stats<<<6, 256, 0, stream>>>(stat);

  fps_kernel<<<BATCH, 256, 0, stream>>>(xyz, nxyz, out);
  query_ball<<<(BATCH * SPTS) / 8, 256, 0, stream>>>(xyz, nxyz, idx);
  build_h0<<<MROWS / 8, 256, 0, stream>>>(xyz, pts, idx, nxyz, h0);

  const int strips = MROWS / 16;                          // 32768
  gemm_wmma<96, 64, 4><<<strips / 4 / 8, 256, 0, stream>>>(h0, wt, b0, y0,
                                                           stat + 0, stat + 128);
  finalize_stats<<<1, 128, 0, stream>>>(stat + 0, stat + 128, g0, be0,
                                        stat + 256, stat + 384, 64);
  bn_relu64<<<(MROWS * 64 / 8) / 256, 256, 0, stream>>>(y0, stat + 256, stat + 384);

  gemm_wmma<64, 64, 4><<<strips / 4 / 8, 256, 0, stream>>>(y0, wt + 8192, b1, y1,
                                                           stat + 512, stat + 640);
  finalize_stats<<<1, 128, 0, stream>>>(stat + 512, stat + 640, g1, be1,
                                        stat + 768, stat + 896, 64);
  bn_relu64<<<(MROWS * 64 / 8) / 256, 256, 0, stream>>>(y1, stat + 768, stat + 896);

  gemm_wmma<64, 128, 2><<<strips / 2 / 8, 256, 0, stream>>>(y1, wt + 16384, b2, y2,
                                                            stat + 1024, stat + 1152);
  finalize_stats<<<1, 128, 0, stream>>>(stat + 1024, stat + 1152, g2, be2,
                                        stat + 1280, stat + 1408, 128);

  final_max<<<(BATCH * SPTS) / 8, 256, 0, stream>>>(y2, stat + 1280, stat + 1408,
                                                    out + BATCH * 3 * SPTS);
}
